// SE3Transformer_2259152797792
// MI455X (gfx1250) — compile-verified, matrix operand order checked by reading, not run on hardware
//
#include <hip/hip_runtime.h>
#include <hip/hip_bf16.h>
#include <string>
#include <vector>
#include <unordered_map>
#include <array>
#include <algorithm>

// ---------------- problem constants ----------------
static const int NN = 4096;     // nodes
static const int EE = 32768;    // edges
static const int CUMD[5] = {0, 1, 4, 9, 16};   // sum of (2k+1)

typedef __attribute__((ext_vector_type(16))) _Float16 v16h;
typedef __attribute__((ext_vector_type(8)))  _Float16 v8h;
typedef __attribute__((ext_vector_type(8)))  float    v8f;

// 16-bit WMMA fragment = two contiguous 8-half (16B) runs along K per lane:
//   half t<8  -> K = (lane/16)*8 + t
//   half t>=8 -> K = 16 + (lane/16)*8 + (t-8)
// So row-major-along-K sources load as two b128s.  B operands use weights
// pre-transposed to column-major so their K axis is contiguous too.
union FragH { v16h v; v8h h[2]; };

// =====================================================================
// Fused radial-MLP (3x WMMA GEMM stages) + SE(3) tensor-product message.
//   h1 = relu(featE @ W1 + b1)            [16e x 32]   (WMMA, K=32)
//   h2 = relu(h1 @ W2 + b2)               [16e x 32]   (WMMA, K=32)
//   R  = h2 @ W3 + b3                     [16e x WOUT] (WMMA, K=32) -> LDS only
//   out[e,o,p] += sum_{i,f} R[e,o,i,f] * sum_q basis[e,p,q,f] * x[src[e],i,q]
// R never touches HBM (saves ~3 GB of traffic vs. materializing coefficients).
// W*t arguments are f16 column-major ([cols][32] / [cols][128]).
// =====================================================================
template <int MI, int MO, int DI, int DO, bool NODE_OUT>
__global__ __launch_bounds__(256) void radial_pair(
    const _Float16* __restrict__ featE, const _Float16* __restrict__ W1t,
    const float* __restrict__ b1, const _Float16* __restrict__ W2t,
    const float* __restrict__ b2, const _Float16* __restrict__ W3t,
    const float* __restrict__ b3, const float* __restrict__ basis,
    const float* __restrict__ xin, const int* __restrict__ srcI,
    const int* __restrict__ dstI, float* __restrict__ outp)
{
    constexpr int F    = 2 * (DI < DO ? DI : DO) + 1;
    constexpr int Q    = 2 * DI + 1;
    constexpr int P    = 2 * DO + 1;
    constexpr int WOUT = F * MI * MO;
    constexpr int TILE = 16;

    __shared__ __align__(16) _Float16 h1h[TILE * 32];
    __shared__ __align__(16) _Float16 h2h[TILE * 32];
    __shared__ __align__(16) _Float16 Rl[TILE * WOUT];   // radial coeffs (LDS only)
    __shared__ float Xl[TILE * MI * Q];                  // gathered src features
    __shared__ float Tl[TILE * MI * F];                  // basis . x  (per p)
    __shared__ float Bl[TILE * Q * F];                   // basis slice (per p)

    const int tid  = threadIdx.x;
    const int lane = tid & 31;
    const int wv   = tid >> 5;                 // 8 wave32s
    const int e0   = blockIdx.x * TILE;
    const int mrow = lane & 15;
    const int hi8  = (lane >> 4) << 3;         // 0 or 8
    const int ch   = lane >> 4;                // K-chunk select

    // ---- stage A: h1 = relu(featE @ W1 + b1) ---- (2 col-tiles -> waves 0,1)
    if (wv < 2) {
        FragH a, b; v8f c = {};
        const v8h* ar = (const v8h*)(featE + (size_t)(e0 + mrow) * 32);
        a.h[0] = ar[ch]; a.h[1] = ar[2 + ch];
        const v8h* br = (const v8h*)(W1t + (size_t)(wv * 16 + mrow) * 32);
        b.h[0] = br[ch]; b.h[1] = br[2 + ch];
        c = __builtin_amdgcn_wmma_f32_16x16x32_f16(false, a.v, false, b.v, (short)0, c, false, false);
#pragma unroll
        for (int r = 0; r < 8; r++) {
            int M = r + hi8, Nc = wv * 16 + mrow;
            float v = c[r] + b1[Nc]; v = v > 0.f ? v : 0.f;
            h1h[M * 32 + Nc] = (_Float16)v;
        }
    }
    __syncthreads();

    // ---- stage B: h2 = relu(h1 @ W2 + b2) ----
    if (wv < 2) {
        FragH a, b; v8f c = {};
        const v8h* ar = (const v8h*)(h1h + mrow * 32);
        a.h[0] = ar[ch]; a.h[1] = ar[2 + ch];
        const v8h* br = (const v8h*)(W2t + (size_t)(wv * 16 + mrow) * 32);
        b.h[0] = br[ch]; b.h[1] = br[2 + ch];
        c = __builtin_amdgcn_wmma_f32_16x16x32_f16(false, a.v, false, b.v, (short)0, c, false, false);
#pragma unroll
        for (int r = 0; r < 8; r++) {
            int M = r + hi8, Nc = wv * 16 + mrow;
            float v = c[r] + b2[Nc]; v = v > 0.f ? v : 0.f;
            h2h[M * 32 + Nc] = (_Float16)v;
        }
    }
    __syncthreads();

    // ---- stage C: R = h2 @ W3 + b3, WOUT/16 col-tiles over 8 waves ----
    {
        FragH a;
        const v8h* ar = (const v8h*)(h2h + mrow * 32);
        a.h[0] = ar[ch]; a.h[1] = ar[2 + ch];
        for (int nt = wv; nt < WOUT / 16; nt += 8) {
            const int col0 = nt * 16;
            // prefetch this wave's next column tile (speculative, safe OOB)
            __builtin_prefetch((const void*)(W3t + (size_t)(col0 + 128 + mrow) * 32), 0, 1);
            FragH b; v8f c = {};
            const v8h* br = (const v8h*)(W3t + (size_t)(col0 + mrow) * 32);
            b.h[0] = br[ch]; b.h[1] = br[2 + ch];
            c = __builtin_amdgcn_wmma_f32_16x16x32_f16(false, a.v, false, b.v, (short)0, c, false, false);
#pragma unroll
            for (int r = 0; r < 8; r++) {
                int M = r + hi8, Nc = col0 + mrow;
                Rl[M * WOUT + Nc] = (_Float16)(c[r] + b3[Nc]);
            }
        }
    }

    // ---- gather x[src] tile ----
    for (int idx = tid; idx < TILE * MI * Q; idx += 256) {
        int e = idx / (MI * Q), rest = idx % (MI * Q);
        Xl[idx] = xin[(size_t)srcI[e0 + e] * (MI * Q) + rest];
    }
    __syncthreads();

    // ---- consume per output order p ----
    const int eg = tid >> 4;   // edge within tile
    const int lt = tid & 15;
    for (int p = 0; p < P; p++) {
        for (int idx = tid; idx < TILE * Q * F; idx += 256) {
            int e = idx / (Q * F), rest = idx % (Q * F);
            Bl[idx] = basis[((size_t)(e0 + e) * P + p) * (Q * F) + rest];
        }
        if (p + 1 < P && lt == 0)
            __builtin_prefetch((const void*)(basis + ((size_t)(e0 + eg) * P + p + 1) * (Q * F)), 0, 1);
        __syncthreads();
        for (int idx = tid; idx < TILE * MI * F; idx += 256) {
            int e = idx / (MI * F), rest = idx % (MI * F);
            int i = rest / F, fi = rest % F;
            float acc = 0.f;
#pragma unroll
            for (int q = 0; q < Q; q++)
                acc += Bl[e * (Q * F) + q * F + fi] * Xl[e * (MI * Q) + i * Q + q];
            Tl[idx] = acc;
        }
        __syncthreads();
        for (int o = lt; o < MO; o += 16) {
            float acc = 0.f;
            for (int i = 0; i < MI; i++)
#pragma unroll
                for (int fi = 0; fi < F; fi++)
                    acc += (float)Rl[eg * WOUT + (o * MI + i) * F + fi] *
                           Tl[eg * (MI * F) + i * F + fi];
            if constexpr (NODE_OUT) {
                atomicAdd(&outp[(size_t)dstI[e0 + eg] * MO + o], acc);
            } else {
                outp[((size_t)(e0 + eg) * MO + o) * P + p] += acc;
            }
        }
        __syncthreads();
    }
}

// =====================================================================
// FC1: [N x 128] @ [128 x 128] + bias, relu -> f16, pure WMMA (K=4x32)
// W1t is f16 column-major [128][128].
// =====================================================================
__global__ __launch_bounds__(256) void fc1_kernel(
    _Float16* __restrict__ midh, const _Float16* __restrict__ xh,
    const _Float16* __restrict__ W1t, const float* __restrict__ b1, int N_)
{
    const int lane = threadIdx.x & 31, wv = threadIdx.x >> 5;
    const int e0 = blockIdx.x * 16, col0 = wv * 16, mrow = lane & 15;
    const int hi8 = (lane >> 4) << 3, ch = lane >> 4;
    const v8h* ar = (const v8h*)(xh + (size_t)(e0 + mrow) * 128);
    const v8h* br = (const v8h*)(W1t + (size_t)(col0 + mrow) * 128);
    v8f c = {};
#pragma unroll
    for (int kk = 0; kk < 4; kk++) {
        FragH a, b;
        a.h[0] = ar[kk * 4 + ch]; a.h[1] = ar[kk * 4 + 2 + ch];
        b.h[0] = br[kk * 4 + ch]; b.h[1] = br[kk * 4 + 2 + ch];
        c = __builtin_amdgcn_wmma_f32_16x16x32_f16(false, a.v, false, b.v, (short)0, c, false, false);
    }
#pragma unroll
    for (int r = 0; r < 8; r++) {
        int M = r + hi8;
        float v = c[r] + b1[col0 + mrow]; v = v > 0.f ? v : 0.f;
        midh[(size_t)(e0 + M) * 128 + col0 + mrow] = (_Float16)v;
    }
}

// ---------------- small VALU kernels ----------------
__global__ void fillk(float* p, float v, int n) {
    int i = blockIdx.x * 256 + threadIdx.x; if (i < n) p[i] = v;
}
__global__ void cvt16(_Float16* d, const float* s, int n) {
    int i = blockIdx.x * 256 + threadIdx.x; if (i < n) d[i] = (_Float16)s[i];
}
// transpose-convert: src [K][C] f32 row-major -> dst [C][K] f16 (column-major of src)
__global__ void cvt16_t(_Float16* d, const float* s, int K, int C) {
    int idx = blockIdx.x * 256 + threadIdx.x; if (idx >= K * C) return;
    int c = idx / K, k = idx % K;
    d[idx] = (_Float16)s[(size_t)k * C + c];
}
// W1 [5][32] -> dst [32 cols][32 rows] f16, rows 5..31 zero
__global__ void cvt_pad_w1_t(_Float16* d, const float* s) {
    int idx = blockIdx.x * 256 + threadIdx.x; if (idx >= 32 * 32) return;
    int c = idx / 32, k = idx % 32;
    d[idx] = (k < 5) ? (_Float16)s[k * 32 + c] : (_Float16)0.f;
}
__global__ void featk(_Float16* fe, const float* w, const float* dv, int E_) {
    int e = blockIdx.x * 256 + threadIdx.x; if (e >= E_) return;
    float x = dv[e * 3], y = dv[e * 3 + 1], z = dv[e * 3 + 2];
    for (int j = 0; j < 4; j++) fe[e * 32 + j] = (_Float16)w[e * 4 + j];
    fe[e * 32 + 4] = (_Float16)sqrtf(x * x + y * y + z * z);
    for (int j = 5; j < 32; j++) fe[e * 32 + j] = (_Float16)0.f;
}
__global__ void qk(float* q, const float* Wq, const float* h, int N_, int P) {
    int idx = blockIdx.x * 256 + threadIdx.x; if (idx >= N_ * 8 * P) return;
    int n = idx / (8 * P), r = idx % (8 * P), o = r / P, pp = r % P;
    float s = 0.f;
    for (int i = 0; i < 32; i++) s += Wq[o * 32 + i] * h[((size_t)n * 32 + i) * P + pp];
    q[idx] = s;
}
__global__ void logit_acc(float* logits, const float* k, const float* q,
                          const int* dst, int P, int E_) {
    int e = blockIdx.x * 256 + threadIdx.x; if (e >= E_) return;
    int d_ = dst[e]; float acc = 0.f;
    for (int i = 0; i < 8; i++)
        for (int pp = 0; pp < P; pp++)
            acc += k[((size_t)e * 8 + i) * P + pp] * q[((size_t)d_ * 8 + i) * P + pp];
    logits[e] += acc;
}
__global__ void scalek(float* p, float s, int n) {
    int i = blockIdx.x * 256 + threadIdx.x; if (i < n) p[i] *= s;
}
__device__ void atomicMaxF(float* a, float v) {
    int* ai = (int*)a; int old = __float_as_int(*a);
    while (__int_as_float(old) < v) {
        int assumed = old;
        old = atomicCAS(ai, assumed, __float_as_int(v));
        if (old == assumed) break;
    }
}
__global__ void amaxk(float* m, const float* logits, const int* dst, int E_) {
    int e = blockIdx.x * 256 + threadIdx.x; if (e < E_) atomicMaxF(&m[dst[e]], logits[e]);
}
__global__ void maxfix(float* m, int n) {
    int i = blockIdx.x * 256 + threadIdx.x; if (i < n && m[i] < -1e29f) m[i] = 0.f;
}
__global__ void expk(float* ev, float* zs, const float* lg, const float* mm,
                     const int* dst, int E_) {
    int e = blockIdx.x * 256 + threadIdx.x; if (e >= E_) return;
    float x = expf(lg[e] - mm[dst[e]]); ev[e] = x; atomicAdd(&zs[dst[e]], x);
}
__global__ void alphak(float* ev, const float* zs, const int* dst, int E_) {
    int e = blockIdx.x * 256 + threadIdx.x; if (e < E_) ev[e] /= fmaxf(zs[dst[e]], 1e-9f);
}
__global__ void aggk(float* z, const float* alpha, const float* v, const int* dst,
                     int P, int E_) {
    int idx = blockIdx.x * 256 + threadIdx.x; if (idx >= E_ * 8 * P) return;
    int e = idx / (8 * P), r = idx % (8 * P);
    atomicAdd(&z[(size_t)dst[e] * (8 * P) + r], alpha[e] * v[idx]);
}
__global__ __launch_bounds__(64) void projnorm(
    float* hout, const float* z, const float* hin, const float* Wp,
    const float* Wn, const float* bn, int P, int CATC)
{
    __shared__ float xl[32 * 7]; __shared__ float nr[32]; __shared__ float sc[32];
    const int n = blockIdx.x, tid = threadIdx.x;
    for (int idx = tid; idx < 32 * P; idx += 64) {
        int o = idx / P, pp = idx % P; float acc = 0.f;
        for (int i = 0; i < CATC; i++) {
            float cv = (i < 8) ? z[((size_t)n * 8 + i) * P + pp]
                               : hin[((size_t)n * 32 + (i - 8)) * P + pp];
            acc += Wp[o * CATC + i] * cv;
        }
        xl[idx] = acc;
    }
    __syncthreads();
    if (tid < 32) {
        float s = 1e-12f;
        for (int pp = 0; pp < P; pp++) { float t = xl[tid * P + pp]; s += t * t; }
        nr[tid] = sqrtf(s);
    }
    __syncthreads();
    if (tid < 32) {
        float s = bn[tid];
        for (int ci = 0; ci < 32; ci++) s += nr[ci] * Wn[ci * 32 + tid];
        sc[tid] = s > 0.f ? s : 0.f;
    }
    __syncthreads();
    for (int idx = tid; idx < 32 * P; idx += 64) {
        int o = idx / P;
        hout[(size_t)n * 32 * P + idx] = sc[o] / nr[o] * xl[idx];
    }
}
__global__ void degk(float* deg, const int* dst, int E_) {
    int e = blockIdx.x * 256 + threadIdx.x; if (e < E_) atomicAdd(&deg[dst[e]], 1.f);
}
__global__ void convfin(float* h0, const float* acc, const float* deg,
                        const float* wself, const float* h0in, int N_) {
    int idx = blockIdx.x * 256 + threadIdx.x; if (idx >= N_ * 128) return;
    int n = idx / 128, o = idx % 128;
    float s = acc[idx] / fmaxf(deg[n], 1.f);
    for (int i = 0; i < 32; i++) s += wself[o * 32 + i] * h0in[(size_t)n * 32 + i];
    h0[idx] = s;
}
__global__ void fc2k(float* out, const _Float16* midh, const float* w2,
                     const float* b2, int N_) {
    int idx = blockIdx.x * 256 + threadIdx.x; if (idx >= N_ * 3) return;
    int n = idx / 3, j = idx % 3;
    float s = b2[j];
    for (int c = 0; c < 128; c++) s += (float)midh[(size_t)n * 128 + c] * w2[c * 3 + j];
    out[idx] = s;
}

// ---------------- template dispatch ----------------
typedef const _Float16* cph; typedef const float* cpf; typedef const int* cpi;
static void rp8(int di, int doo, dim3 g, hipStream_t st, cph fe, cph w1, cpf b1,
                cph w2, cpf b2, cph w3, cpf b3, cpf bs, cpf xin, cpi sr, cpi ds, float* out) {
    switch (di * 4 + doo) {
#define C8(DI_, DO_) case DI_ * 4 + DO_: \
    radial_pair<32, 8, DI_, DO_, false><<<g, 256, 0, st>>>(fe, w1, b1, w2, b2, w3, b3, bs, xin, sr, ds, out); break;
    C8(0,0) C8(0,1) C8(0,2) C8(0,3) C8(1,0) C8(1,1) C8(1,2) C8(1,3)
    C8(2,0) C8(2,1) C8(2,2) C8(2,3) C8(3,0) C8(3,1) C8(3,2) C8(3,3)
#undef C8
    }
}
static void rpc(int di, dim3 g, hipStream_t st, cph fe, cph w1, cpf b1, cph w2, cpf b2,
                cph w3, cpf b3, cpf bs, cpf xin, cpi sr, cpi ds, float* out) {
    switch (di) {
#define CC(DI_) case DI_: \
    radial_pair<32, 128, DI_, 0, true><<<g, 256, 0, st>>>(fe, w1, b1, w2, b2, w3, b3, bs, xin, sr, ds, out); break;
    CC(0) CC(1) CC(2) CC(3)
#undef CC
    }
}

// ---------------- input-order reconstruction ----------------
static void build_order(bool sorted, std::vector<std::string>& o) {
    auto rad = [&](const std::string& p) {
        if (sorted) { o.push_back(p+".W1"); o.push_back(p+".W2"); o.push_back(p+".W3");
                      o.push_back(p+".b1"); o.push_back(p+".b2"); o.push_back(p+".b3"); }
        else        { o.push_back(p+".W1"); o.push_back(p+".b1"); o.push_back(p+".W2");
                      o.push_back(p+".b2"); o.push_back(p+".W3"); o.push_back(p+".b3"); }
    };
    auto basis_all = [&]() {
        for (int di = 0; di < 4; di++) for (int dd = 0; dd < 4; dd++)
            o.push_back("basis" + std::to_string(di) + std::to_string(dd));
    };
    auto vk = [&](int l, const char* tag, int nqdo) {
        int ndin = l == 0 ? 1 : 4;
        std::string L = "l" + std::to_string(l) + "." + tag;
        if (sorted) { for (int di = 0; di < ndin; di++) for (int dd = 0; dd < nqdo; dd++)
                          rad(L + std::to_string(di) + std::to_string(dd)); }
        else        { for (int dd = 0; dd < nqdo; dd++) for (int di = 0; di < ndin; di++)
                          rad(L + std::to_string(di) + std::to_string(dd)); }
    };
    auto qs = [&](int l, int nq) { for (int dd = 0; dd < nq; dd++)
        o.push_back("l" + std::to_string(l) + ".q" + std::to_string(dd)); };
    auto projs = [&](int l) { for (int dd = 0; dd < 4; dd++)
        o.push_back("l" + std::to_string(l) + ".proj" + std::to_string(dd)); };
    auto norms = [&](int l) { for (int dd = 0; dd < 4; dd++) {
        o.push_back("l" + std::to_string(l) + ".norm" + std::to_string(dd) + ".W");
        o.push_back("l" + std::to_string(l) + ".norm" + std::to_string(dd) + ".b"); } };
    auto layer = [&](int l) {
        int nq = l == 0 ? 1 : 4;
        if (sorted) { vk(l, "k", nq); norms(l); projs(l); qs(l, nq); vk(l, "v", 4); }
        else        { vk(l, "v", 4); vk(l, "k", nq); qs(l, nq); projs(l); norms(l); }
    };
    auto conv = [&]() { for (int di = 0; di < 4; di++) rad("conv" + std::to_string(di));
                        o.push_back("conv.self"); };
    auto fc = [&]() {
        if (sorted) { o.push_back("fc.W1"); o.push_back("fc.W2");
                      o.push_back("fc.b1"); o.push_back("fc.b2"); }
        else        { o.push_back("fc.W1"); o.push_back("fc.b1");
                      o.push_back("fc.W2"); o.push_back("fc.b2"); }
    };
    if (sorted) {
        basis_all(); o.push_back("d"); o.push_back("dst"); o.push_back("f");
        conv(); fc(); layer(0); layer(1);
        o.push_back("src"); o.push_back("w");
    } else {
        o.push_back("f"); o.push_back("w"); o.push_back("d");
        o.push_back("src"); o.push_back("dst");
        basis_all(); layer(0); layer(1); conv(); fc();
    }
}

extern "C" void kernel_launch(void* const* d_in, const int* in_sizes, int n_in,
                              void* d_out, int out_size, void* d_ws, size_t ws_size,
                              hipStream_t stream) {
    (void)n_in; (void)out_size; (void)ws_size;
    // detect flatten mode: sorted(jax-tree) puts basis(0,0) [E]=32768 first;
    // insertion puts f [N*32]=131072 first.
    const bool sorted_mode = (in_sizes[0] == EE);
    std::vector<std::string> order; build_order(sorted_mode, order);
    std::unordered_map<std::string, int> ix;
    for (int i = 0; i < (int)order.size(); i++) ix[order[i]] = i;
    auto getf = [&](const std::string& s) { return (const float*)d_in[ix[s]]; };
    auto geti = [&](const std::string& s) { return (const int*)d_in[ix[s]]; };

    const int* srcP = geti("src");
    const int* dstP = geti("dst");

    // ---- workspace bump allocator ----
    char* W = (char*)d_ws; size_t off = 0;
    auto alloc = [&](size_t b) { size_t o = off; off = (off + b + 255) & ~(size_t)255; return W + o; };
    _Float16* featE  = (_Float16*)alloc((size_t)EE * 32 * 2);
    float* vbuf      = (float*)alloc((size_t)EE * 8 * 16 * 4);
    float* kbuf      = (float*)alloc((size_t)EE * 8 * 16 * 4);
    float* qbuf      = (float*)alloc((size_t)NN * 8 * 16 * 4);
    float* hA        = (float*)alloc((size_t)NN * 32 * 16 * 4);
    float* hB        = (float*)alloc((size_t)NN * 32 * 16 * 4);
    float* logits    = (float*)alloc((size_t)EE * 4);
    float* expv      = (float*)alloc((size_t)EE * 4);
    float* mmax      = (float*)alloc((size_t)NN * 4);
    float* zsum      = (float*)alloc((size_t)NN * 4);
    float* aggz      = (float*)alloc((size_t)NN * 8 * 16 * 4);
    float* convacc   = (float*)alloc((size_t)NN * 128 * 4);
    float* degb      = (float*)alloc((size_t)NN * 4);
    float* h0f       = (float*)alloc((size_t)NN * 128 * 4);
    _Float16* xh     = (_Float16*)alloc((size_t)NN * 128 * 2);
    _Float16* midh   = (_Float16*)alloc((size_t)NN * 128 * 2);

    auto g1 = [](int n) { return dim3((n + 255) / 256); };
    auto fill = [&](float* p, float v, int n) { fillk<<<g1(n), 256, 0, stream>>>(p, v, n); };
    auto hptr = [&](float* base, int dd) { return base + (size_t)NN * 32 * CUMD[dd]; };
    auto vptr = [&](float* base, int dd) { return base + (size_t)EE * 8 * CUMD[dd]; };
    auto qptr = [&](int dd) { return qbuf + (size_t)NN * 8 * CUMD[dd]; };
    auto zptr = [&](int dd) { return aggz + (size_t)NN * 8 * CUMD[dd]; };

    // prep: weights -> f16, TRANSPOSED (column-major) so WMMA B-fragments
    // are two contiguous b128 loads per lane.  W1 zero-padded 5->32 rows.
    auto prep_net = [&](const std::string& pre, int wout) {
        _Float16* w1t = (_Float16*)alloc(32 * 32 * 2);
        _Float16* w2t = (_Float16*)alloc(32 * 32 * 2);
        _Float16* w3t = (_Float16*)alloc((size_t)32 * wout * 2);
        cvt_pad_w1_t<<<g1(1024), 256, 0, stream>>>(w1t, getf(pre + ".W1"));
        cvt16_t<<<g1(1024), 256, 0, stream>>>(w2t, getf(pre + ".W2"), 32, 32);
        cvt16_t<<<g1(32 * wout), 256, 0, stream>>>(w3t, getf(pre + ".W3"), 32, wout);
        return std::array<_Float16*, 3>{w1t, w2t, w3t};
    };
    auto run8 = [&](const std::string& pre, int di, int doo, const float* xin, float* outp) {
        int F = 2 * std::min(di, doo) + 1, wout = F * 32 * 8;
        auto wn = prep_net(pre, wout);
        rp8(di, doo, dim3(EE / 16), stream, featE, wn[0], getf(pre + ".b1"),
            wn[1], getf(pre + ".b2"), wn[2], getf(pre + ".b3"),
            getf("basis" + std::to_string(di) + std::to_string(doo)),
            xin, srcP, dstP, outp);
    };

    // ---- edge features ----
    featk<<<g1(EE), 256, 0, stream>>>(featE, getf("w"), getf("d"), EE);

    // ---- attention layers ----
    for (int l = 0; l < 2; l++) {
        const int ndin = l ? 4 : 1, nq = l ? 4 : 1;
        float* houtb = l ? hB : hA;
        auto xin_of = [&](int di) { return l ? (const float*)hptr(hA, di) : getf("f"); };

        fill(vbuf, 0.f, EE * 8 * 16);
        fill(kbuf, 0.f, EE * 8 * 16);
        for (int doo = 0; doo < 4; doo++)
            for (int di = 0; di < ndin; di++)
                run8("l" + std::to_string(l) + ".v" + std::to_string(di) + std::to_string(doo),
                     di, doo, xin_of(di), vptr(vbuf, doo));
        for (int doo = 0; doo < nq; doo++)
            for (int di = 0; di < ndin; di++)
                run8("l" + std::to_string(l) + ".k" + std::to_string(di) + std::to_string(doo),
                     di, doo, xin_of(di), vptr(kbuf, doo));
        for (int dd = 0; dd < nq; dd++) {
            int P = 2 * dd + 1;
            qk<<<g1(NN * 8 * P), 256, 0, stream>>>(
                qptr(dd), getf("l" + std::to_string(l) + ".q" + std::to_string(dd)),
                xin_of(dd), NN, P);
        }
        fill(logits, 0.f, EE);
        for (int dd = 0; dd < nq; dd++)
            logit_acc<<<g1(EE), 256, 0, stream>>>(logits, vptr(kbuf, dd), qptr(dd), dstP,
                                                  2 * dd + 1, EE);
        scalek<<<g1(EE), 256, 0, stream>>>(logits, l ? 0.08838834764f : 0.35355339059f, EE);
        // segment softmax over dst
        fill(mmax, -1e30f, NN);
        amaxk<<<g1(EE), 256, 0, stream>>>(mmax, logits, dstP, EE);
        maxfix<<<g1(NN), 256, 0, stream>>>(mmax, NN);
        fill(zsum, 0.f, NN);
        expk<<<g1(EE), 256, 0, stream>>>(expv, zsum, logits, mmax, dstP, EE);
        alphak<<<g1(EE), 256, 0, stream>>>(expv, zsum, dstP, EE);
        // aggregate alpha*v -> nodes, then proj + norm nonlinearity
        for (int dd = 0; dd < 4; dd++) {
            int P = 2 * dd + 1;
            fill(zptr(dd), 0.f, NN * 8 * P);
            aggk<<<g1(EE * 8 * P), 256, 0, stream>>>(zptr(dd), expv, vptr(vbuf, dd), dstP, P, EE);
        }
        for (int dd = 0; dd < 4; dd++) {
            int P = 2 * dd + 1;
            int CATC = (l || dd == 0) ? 40 : 8;
            const float* hinp = l ? (const float*)hptr(hA, dd)
                                  : (dd == 0 ? getf("f") : (const float*)nullptr);
            std::string L = "l" + std::to_string(l);
            projnorm<<<NN, 64, 0, stream>>>(
                hptr(houtb, dd), zptr(dd), hinp,
                getf(L + ".proj" + std::to_string(dd)),
                getf(L + ".norm" + std::to_string(dd) + ".W"),
                getf(L + ".norm" + std::to_string(dd) + ".b"), P, CATC);
        }
    }

    // ---- final conv -> degree 0 (mean aggregation, fused radial+TP, atomic to nodes) ----
    fill(convacc, 0.f, NN * 128);
    fill(degb, 0.f, NN);
    degk<<<g1(EE), 256, 0, stream>>>(degb, dstP, EE);
    for (int di = 0; di < 4; di++) {
        std::string pre = "conv" + std::to_string(di);
        auto wn = prep_net(pre, 32 * 128);
        rpc(di, dim3(EE / 16), stream, featE, wn[0], getf(pre + ".b1"),
            wn[1], getf(pre + ".b2"), wn[2], getf(pre + ".b3"),
            getf("basis" + std::to_string(di) + "0"), hptr(hB, di), srcP, dstP, convacc);
    }
    convfin<<<g1(NN * 128), 256, 0, stream>>>(h0f, convacc, degb, getf("conv.self"),
                                              hptr(hB, 0), NN);

    // ---- output MLP: WMMA 4096x128x128 then 128->3 ----
    cvt16<<<g1(NN * 128), 256, 0, stream>>>(xh, h0f, NN * 128);
    _Float16* fw1t = (_Float16*)alloc(128 * 128 * 2);
    cvt16_t<<<g1(128 * 128), 256, 0, stream>>>(fw1t, getf("fc.W1"), 128, 128);
    fc1_kernel<<<NN / 16, 256, 0, stream>>>(midh, xh, fw1t, getf("fc.b1"), NN);
    fc2k<<<g1(NN * 3), 256, 0, stream>>>((float*)d_out, midh, getf("fc.W2"), getf("fc.b2"), NN);
}
